// AWD_LSTM_60799557042284
// MI455X (gfx1250) — matrix operand order repaired and checked by reading, not compiled, not run on hardware
//
#include <hip/hip_runtime.h>

// ---------------- dims ----------------
#define T_STEPS 128
#define BATCH   64
#define EMB     400
#define EMBP    416      // EMB padded to mult of 32 (K dim for WMMA)
#define HID     1152
#define NTOK    33278
#define NTOKP   33280    // padded to mult of 64
#define G0      4608     // 4*HID
#define G2      1600     // 4*EMB
#define ROWS    (T_STEPS*BATCH)   // 8192

typedef __attribute__((ext_vector_type(16))) __bf16 v16bf;
typedef __attribute__((ext_vector_type(8)))  float  v8f;

union AFrag { v16bf v; uint4 q[2]; };

// A-matrix 16x32 bf16 fragment (row-major A[M][lda], tile at m0,k0)
// lane l: m = m0+(l&15); halves 0..7 -> k0+(l>>4)*8+{0..7}; halves 8..15 -> k0+16+(l>>4)*8+{0..7}
__device__ __forceinline__ v16bf load_a_frag(const __bf16* __restrict__ A, int lda,
                                             int m0, int k0, int lane) {
  int m = m0 + (lane & 15);
  int g = lane >> 4;
  const __bf16* p = A + (long)m * lda + k0 + g * 8;
  AFrag f;
  f.q[0] = *(const uint4*)(p);
  f.q[1] = *(const uint4*)(p + 16);
  return f.v;
}

// B-matrix 32x16 bf16 fragment for A @ W^T (W row-major [N][ldw], tile at n0,k0)
// lane l: n = n0+(l&15); halves 0..7 -> k0+(l>>4)*8+{0..7}; halves 8..15 -> k0+16+(l>>4)*8+{0..7}
__device__ __forceinline__ v16bf load_b_frag(const __bf16* __restrict__ W, int ldw,
                                             int n0, int k0, int lane) {
  int n = n0 + (lane & 15);
  int g = lane >> 4;
  const __bf16* p = W + (long)n * ldw + k0 + g * 8;
  AFrag f;
  f.q[0] = *(const uint4*)(p);
  f.q[1] = *(const uint4*)(p + 16);
  return f.v;
}

__device__ __forceinline__ v8f wmma_bf16(v16bf a, v16bf b, v8f c) {
  return __builtin_amdgcn_wmma_f32_16x16x32_bf16(false, a, false, b, (short)0, c, false, false);
}

__device__ __forceinline__ float sigm(float x) { return 1.0f / (1.0f + __expf(-x)); }
__device__ __forceinline__ float tanh_fast(float x) {
  float t = __expf(-2.0f * fabsf(x));
  float r = (1.0f - t) / (1.0f + t);
  return x < 0.0f ? -r : r;
}

// ---------------- fp32 -> bf16 convert with zero padding ----------------
__global__ void k_cvt_pad(const float* __restrict__ src, __bf16* __restrict__ dst,
                          int srows, int scols, int drows, int dcols) {
  long i = (long)blockIdx.x * blockDim.x + threadIdx.x;
  long tot = (long)drows * dcols;
  if (i >= tot) return;
  int r = (int)(i / dcols);
  int c = (int)(i - (long)r * dcols);
  float v = (r < srows && c < scols) ? src[(long)r * scols + c] : 0.0f;
  dst[i] = (__bf16)v;
}

__global__ void k_copy_f32(const float* __restrict__ src, float* __restrict__ dst, int n) {
  int i = blockIdx.x * blockDim.x + threadIdx.x;
  if (i < n) dst[i] = src[i];
}

// ---------------- generic WMMA GEMM: out = A @ W^T (+ b1 + b2) ----------------
// one wave (32 thr) computes a 16(M) x 64(N) tile; grid = (Npad/64, M/16)
__global__ void k_gemm_bias(const __bf16* __restrict__ A, int lda,
                            const __bf16* __restrict__ W, int ldw,
                            const float* __restrict__ b1, const float* __restrict__ b2,
                            float* __restrict__ out, long ldOut, int Nstore, int K) {
  int lane = threadIdx.x;
  int m0 = blockIdx.y * 16;
  int n0 = blockIdx.x * 64;
  v8f acc[4];
#pragma unroll
  for (int s = 0; s < 4; s++)
#pragma unroll
    for (int v = 0; v < 8; v++) acc[s][v] = 0.0f;

  for (int k0 = 0; k0 < K; k0 += 32) {
    v16bf a = load_a_frag(A, lda, m0, k0, lane);
#pragma unroll
    for (int s = 0; s < 4; s++) {
      v16bf b = load_b_frag(W, ldw, n0 + 16 * s, k0, lane);
      acc[s] = wmma_bf16(a, b, acc[s]);
    }
  }
  int g = lane >> 4;
#pragma unroll
  for (int s = 0; s < 4; s++) {
    int n = n0 + 16 * s + (lane & 15);
    if (n < Nstore) {
      float bias = 0.0f;
      if (b1) bias += b1[n];
      if (b2) bias += b2[n];
#pragma unroll
      for (int v = 0; v < 8; v++)
        out[(long)(m0 + g * 8 + v) * ldOut + n] = acc[s][v] + bias;
    }
  }
}

// ---------------- fused LSTM step (one layer, one timestep) ----------------
// wave computes 16(batch) x 16(hidden) block of all 4 gates, then the cell update.
// grid = (dout/16, BATCH/16), block = 32
__global__ void k_lstm_step(
    const float* __restrict__ pre, int preLd, int preRowBase,          // layer0: precomputed x-gates(+bias)
    const float* __restrict__ b_ih, const float* __restrict__ b_hh,    // layers 1/2: biases
    const __bf16* __restrict__ X, int ldx, int K1,                     // lower-layer h (input GEMM)
    const __bf16* __restrict__ Wih,
    const __bf16* __restrict__ H, int ldh, int K2,                     // own h_{t-1} (recurrent GEMM)
    const __bf16* __restrict__ Whh,
    float* __restrict__ C,                                             // [BATCH][dout] fp32 cell state
    __bf16* __restrict__ Hout, int ldho,                               // new h (double-buffered)
    __bf16* __restrict__ Hist, int ldhist, int histRowBase,            // layer2: history for decoder
    int dout) {
  int lane = threadIdx.x;
  int nt = blockIdx.x;
  int m0 = blockIdx.y * 16;
  int lane15 = lane & 15, g = lane >> 4;
  int n = nt * 16 + lane15;

  v8f acc[4];
  if (pre) {
#pragma unroll
    for (int s = 0; s < 4; s++) {
      const float* p = pre + (long)(preRowBase + m0 + g * 8) * preLd + s * dout + n;
#pragma unroll
      for (int v = 0; v < 8; v++) acc[s][v] = p[(long)v * preLd];
    }
  } else {
#pragma unroll
    for (int s = 0; s < 4; s++) {
      float bias = b_ih[s * dout + n] + b_hh[s * dout + n];
#pragma unroll
      for (int v = 0; v < 8; v++) acc[s][v] = bias;
    }
  }

  for (int k0 = 0; k0 < K1; k0 += 32) {
    v16bf a = load_a_frag(X, ldx, m0, k0, lane);
#pragma unroll
    for (int s = 0; s < 4; s++) {
      v16bf b = load_b_frag(Wih, K1, s * dout + nt * 16, k0, lane);
      acc[s] = wmma_bf16(a, b, acc[s]);
    }
  }
  for (int k0 = 0; k0 < K2; k0 += 32) {
    v16bf a = load_a_frag(H, ldh, m0, k0, lane);
#pragma unroll
    for (int s = 0; s < 4; s++) {
      v16bf b = load_b_frag(Whh, K2, s * dout + nt * 16, k0, lane);
      acc[s] = wmma_bf16(a, b, acc[s]);
    }
  }

  // in-register cell update on accumulator layout: m = m0+g*8+v, col n
#pragma unroll
  for (int v = 0; v < 8; v++) {
    int m = m0 + g * 8 + v;
    long ci = (long)m * dout + n;
    float cold = C[ci];
    float iv = sigm(acc[0][v]);
    float fv = sigm(acc[1][v]);
    float gv = tanh_fast(acc[2][v]);
    float ov = sigm(acc[3][v]);
    float cnew = fv * cold + iv * gv;
    float h = ov * tanh_fast(cnew);
    C[ci] = cnew;
    __bf16 hb = (__bf16)h;
    Hout[(long)m * ldho + n] = hb;
    if (Hist) Hist[(long)(histRowBase + m) * ldhist + n] = hb;
  }
}

// ---------------- host side ----------------
extern "C" void kernel_launch(void* const* d_in, const int* in_sizes, int n_in,
                              void* d_out, int out_size, void* d_ws, size_t ws_size,
                              hipStream_t stream) {
  const float* x    = (const float*)d_in[0];
  const float* h00  = (const float*)d_in[1];
  const float* h01  = (const float*)d_in[2];
  const float* h02  = (const float*)d_in[3];
  const float* c00  = (const float*)d_in[4];
  const float* c01  = (const float*)d_in[5];
  const float* c02  = (const float*)d_in[6];
  const float* Wih0 = (const float*)d_in[7];
  const float* Whh0 = (const float*)d_in[8];
  const float* bih0 = (const float*)d_in[9];
  const float* bhh0 = (const float*)d_in[10];
  const float* Wih1 = (const float*)d_in[11];
  const float* Whh1 = (const float*)d_in[12];
  const float* bih1 = (const float*)d_in[13];
  const float* bhh1 = (const float*)d_in[14];
  const float* Wih2 = (const float*)d_in[15];
  const float* Whh2 = (const float*)d_in[16];
  const float* bih2 = (const float*)d_in[17];
  const float* bhh2 = (const float*)d_in[18];
  const float* Wdec = (const float*)d_in[19];
  const float* bdec = (const float*)d_in[20];
  float* out = (float*)d_out;

  // workspace carve-up (256B aligned)
  char* cur = (char*)d_ws;
  auto alloc = [&](size_t bytes) -> void* {
    void* p = (void*)cur;
    cur += (bytes + 255) & ~(size_t)255;
    return p;
  };
  __bf16* Wih0p = (__bf16*)alloc((size_t)G0 * EMBP * 2);
  __bf16* Whh0b = (__bf16*)alloc((size_t)G0 * HID * 2);
  __bf16* Wih1b = (__bf16*)alloc((size_t)G0 * HID * 2);
  __bf16* Whh1b = (__bf16*)alloc((size_t)G0 * HID * 2);
  __bf16* Wih2b = (__bf16*)alloc((size_t)G2 * HID * 2);
  __bf16* Whh2p = (__bf16*)alloc((size_t)G2 * EMBP * 2);
  __bf16* Wdecp = (__bf16*)alloc((size_t)NTOKP * EMBP * 2);
  __bf16* xbf   = (__bf16*)alloc((size_t)ROWS * EMBP * 2);
  __bf16* hist  = (__bf16*)alloc((size_t)ROWS * EMBP * 2);
  float*  pre0  = (float*) alloc((size_t)ROWS * G0 * 4);
  __bf16* h0b[2] = { (__bf16*)alloc((size_t)BATCH * HID * 2),
                     (__bf16*)alloc((size_t)BATCH * HID * 2) };
  __bf16* h1b[2] = { (__bf16*)alloc((size_t)BATCH * HID * 2),
                     (__bf16*)alloc((size_t)BATCH * HID * 2) };
  __bf16* h2b[2] = { (__bf16*)alloc((size_t)BATCH * EMBP * 2),
                     (__bf16*)alloc((size_t)BATCH * EMBP * 2) };
  float* c0 = (float*)alloc((size_t)BATCH * HID * 4);
  float* c1 = (float*)alloc((size_t)BATCH * HID * 4);
  float* c2 = (float*)alloc((size_t)BATCH * EMB * 4);

  auto cvt = [&](const float* src, __bf16* dst, int sr, int sc, int dr, int dc) {
    long tot = (long)dr * dc;
    int blocks = (int)((tot + 255) / 256);
    k_cvt_pad<<<blocks, 256, 0, stream>>>(src, dst, sr, sc, dr, dc);
  };

  // weight / input conversion (once per call)
  cvt(Wih0, Wih0p, G0, EMB, G0, EMBP);
  cvt(Whh0, Whh0b, G0, HID, G0, HID);
  cvt(Wih1, Wih1b, G0, HID, G0, HID);
  cvt(Whh1, Whh1b, G0, HID, G0, HID);
  cvt(Wih2, Wih2b, G2, HID, G2, HID);
  cvt(Whh2, Whh2p, G2, EMB, G2, EMBP);
  cvt(Wdec, Wdecp, NTOK, EMB, NTOKP, EMBP);
  cvt(x, xbf, ROWS, EMB, ROWS, EMBP);
  cvt(x, hist, 0, 0, ROWS, EMBP);  // zero-fill history (pad cols must be 0)

  // initial hidden / cell state (both h ping-pong buffers so pads stay zero)
  cvt(h00, h0b[0], BATCH, HID, BATCH, HID);
  cvt(h00, h0b[1], BATCH, HID, BATCH, HID);
  cvt(h01, h1b[0], BATCH, HID, BATCH, HID);
  cvt(h01, h1b[1], BATCH, HID, BATCH, HID);
  cvt(h02, h2b[0], BATCH, EMB, BATCH, EMBP);
  cvt(h02, h2b[1], BATCH, EMB, BATCH, EMBP);
  k_copy_f32<<<(BATCH * HID + 255) / 256, 256, 0, stream>>>(c00, c0, BATCH * HID);
  k_copy_f32<<<(BATCH * HID + 255) / 256, 256, 0, stream>>>(c01, c1, BATCH * HID);
  k_copy_f32<<<(BATCH * EMB + 255) / 256, 256, 0, stream>>>(c02, c2, BATCH * EMB);

  // pre0 = x @ Wih0^T + (b_ih0 + b_hh0)  for all T at once
  k_gemm_bias<<<dim3(G0 / 64, ROWS / 16), 32, 0, stream>>>(
      xbf, EMBP, Wih0p, EMBP, bih0, bhh0, pre0, (long)G0, G0, EMBP);

  // sequential recurrence
  for (int t = 0; t < T_STEPS; t++) {
    int o = t & 1, nw = o ^ 1;
    // layer 0: acc init from pre0, recurrent GEMM only
    k_lstm_step<<<dim3(HID / 16, BATCH / 16), 32, 0, stream>>>(
        pre0, G0, t * BATCH, nullptr, nullptr,
        nullptr, 0, 0, nullptr,
        h0b[o], HID, HID, Whh0b,
        c0, h0b[nw], HID, nullptr, 0, 0, HID);
    // layer 1
    k_lstm_step<<<dim3(HID / 16, BATCH / 16), 32, 0, stream>>>(
        nullptr, 0, 0, bih1, bhh1,
        h0b[nw], HID, HID, Wih1b,
        h1b[o], HID, HID, Whh1b,
        c1, h1b[nw], HID, nullptr, 0, 0, HID);
    // layer 2 (writes decoder history)
    k_lstm_step<<<dim3(EMB / 16, BATCH / 16), 32, 0, stream>>>(
        nullptr, 0, 0, bih2, bhh2,
        h1b[nw], HID, HID, Wih2b,
        h2b[o], EMBP, EMBP, Whh2p,
        c2, h2b[nw], EMBP, hist, EMBP, t * BATCH, EMB);
  }

  // decoder: out = hist @ Wdec^T + b_dec   (store-guarded at N=33278)
  k_gemm_bias<<<dim3(NTOKP / 64, ROWS / 16), 32, 0, stream>>>(
      hist, EMBP, Wdecp, EMBP, bdec, nullptr, out, (long)NTOK, NTOK, EMBP);
}
